// GlobalMambaBlockFromMamba_89756226551980
// MI455X (gfx1250) — matrix operand order repaired and checked by reading, not compiled, not run on hardware
//
#include <hip/hip_runtime.h>
#include <hip/hip_bf16.h>

typedef __attribute__((ext_vector_type(16))) _Float16 v16h;
typedef __attribute__((ext_vector_type(8)))  _Float16 v8h;
typedef __attribute__((ext_vector_type(8)))  float    v8f;

#define DIM      1024
#define D_STATE  16
#define D_CONV   4
#define D_INNER  2048
#define DT_RANK  64
#define BB       2
#define LL       1024
#define NROW     (BB * LL)   // 2048

// ---------------- CDNA5 async copy (global -> LDS), ASYNCcnt-tracked --------
__device__ __forceinline__ void async_b128(unsigned int lds_addr, const void* gptr) {
    asm volatile("global_load_async_to_lds_b128 %0, %1, off"
                 :: "v"(lds_addr), "v"((unsigned long long)(uintptr_t)gptr)
                 : "memory");
}
#if defined(__has_builtin)
#if __has_builtin(__builtin_amdgcn_s_wait_asynccnt)
#define WAIT_ASYNC(n) __builtin_amdgcn_s_wait_asynccnt(n)
#endif
#endif
#ifndef WAIT_ASYNC
#define WAIT_ASYNC(n) asm volatile("s_wait_asynccnt %0" :: "i"(n) : "memory")
#endif

// ------------------------------------------------ weight transpose f32->f16
// src [R][C] row-major f32  ->  dst [Cpad][R] row-major f16 (zero pad C..Cpad)
__global__ void transp_f32_to_f16(const float* __restrict__ src, int R, int C,
                                  _Float16* __restrict__ dst) {
    __shared__ float tile[32][33];
    const int c0 = blockIdx.x * 32, r0 = blockIdx.y * 32;
    const int tx = threadIdx.x, ty = threadIdx.y;
#pragma unroll
    for (int k = 0; k < 4; ++k) {
        const int r = r0 + ty + 8 * k, c = c0 + tx;
        tile[ty + 8 * k][tx] = (c < C) ? src[(size_t)r * C + c] : 0.0f;
    }
    __syncthreads();
#pragma unroll
    for (int k = 0; k < 4; ++k) {
        const int c = c0 + ty + 8 * k, r = r0 + tx;
        dst[(size_t)c * R + r] = (_Float16)tile[tx][ty + 8 * k];
    }
}

// ---------------------------------------------------------------- layernorm
__global__ void ln_kernel(const float* __restrict__ x,
                          const float* __restrict__ w,
                          const float* __restrict__ b,
                          _Float16* __restrict__ xn_h) {
    __shared__ float s_sum[256];
    __shared__ float s_sq[256];
    const int row = blockIdx.x;
    const int tid = threadIdx.x;
    const float4 v = *(const float4*)(x + (size_t)row * DIM + tid * 4);
    s_sum[tid] = v.x + v.y + v.z + v.w;
    s_sq[tid]  = v.x * v.x + v.y * v.y + v.z * v.z + v.w * v.w;
    __syncthreads();
    for (int off = 128; off > 0; off >>= 1) {
        if (tid < off) { s_sum[tid] += s_sum[tid + off]; s_sq[tid] += s_sq[tid + off]; }
        __syncthreads();
    }
    const float mu  = s_sum[0] * (1.0f / DIM);
    const float var = s_sq[0] * (1.0f / DIM) - mu * mu;
    const float rs  = rsqrtf(var + 1e-5f);
    _Float16* o = xn_h + (size_t)row * DIM + tid * 4;
    o[0] = (_Float16)((v.x - mu) * rs * w[tid*4+0] + b[tid*4+0]);
    o[1] = (_Float16)((v.y - mu) * rs * w[tid*4+1] + b[tid*4+1]);
    o[2] = (_Float16)((v.z - mu) * rs * w[tid*4+2] + b[tid*4+2]);
    o[3] = (_Float16)((v.w - mu) * rs * w[tid*4+3] + b[tid*4+3]);
}

// ------------------------------------------------- WMMA f16 GEMM, async-LDS
// C[M,N](ldc,f32) = A_h[M,K](lda) @ WT_h[N][K](ldb=K) + bias ; optional f16 copy Ch.
// EPI: 0=none, 1=softplus. block=256 (8 waves), tile 128x64, K-chunk 64,
// double-buffered global_load_async_to_lds_b128 (6 copies/wave/chunk).
template <int EPI>
__global__ __launch_bounds__(256)
void gemm_wmma(const _Float16* __restrict__ A, int lda,
               const _Float16* __restrict__ WT, int ldb,
               const float* __restrict__ bias,
               float* __restrict__ C, int ldc,
               _Float16* __restrict__ Ch, int ldch,
               int M, int N, int K) {
    __shared__ _Float16 Asub[2 * 128 * 64];
    __shared__ _Float16 Bsub[2 * 64 * 64];

    const int tid  = threadIdx.x;
    const int lane = tid & 31;
    const int wave = tid >> 5;
    const int wm   = wave >> 1;              // 0..3
    const int wn   = wave & 1;               // 0..1
    const int gm0  = blockIdx.y * 128;
    const int n0   = blockIdx.x * 64;
    const int l15  = lane & 15;
    const int kbA  = (lane < 16) ? 0 : 8;
    const int kbB  = (lane < 16) ? 0 : 8;

    const unsigned int aLds0 = (unsigned int)(uintptr_t)&Asub[0];
    const unsigned int bLds0 = (unsigned int)(uintptr_t)&Bsub[0];

    // per-thread copy indices (uniform across chunks)
    const int arow0 = tid >> 3, ac0 = tid & 7;      // + r*32 rows
    const int brow0 = tid >> 3, bc0 = tid & 7;      // + r*32 rows

    auto issue = [&](int kk, int bsel) {
        const unsigned int aL = aLds0 + (unsigned int)(bsel * 128 * 64 * 2);
        const unsigned int bL = bLds0 + (unsigned int)(bsel * 64 * 64 * 2);
#pragma unroll
        for (int r = 0; r < 4; ++r) {
            const int arow = arow0 + r * 32, ac = ac0;
            async_b128(aL + (unsigned int)((arow * 64 + ac * 8) * 2),
                       A + (size_t)(gm0 + arow) * lda + kk + ac * 8);
        }
#pragma unroll
        for (int r = 0; r < 2; ++r) {
            const int brow = brow0 + r * 32, bc = bc0;
            async_b128(bL + (unsigned int)((brow * 64 + bc * 8) * 2),
                       WT + (size_t)(n0 + brow) * ldb + kk + bc * 8);
        }
    };

    v8f acc[4];
#pragma unroll
    for (int i = 0; i < 4; ++i) acc[i] = (v8f)(0.0f);

    const int nch = K >> 6;                  // K % 64 == 0 for all launches
    issue(0, 0);
    if (nch > 1) issue(64, 1);

    for (int i = 0; i < nch; ++i) {
        if (i + 1 < nch) { WAIT_ASYNC(6); } else { WAIT_ASYNC(0); }
        __syncthreads();

        const _Float16* Ab = &Asub[(i & 1) * 128 * 64];
        const _Float16* Bb = &Bsub[(i & 1) * 64 * 64];
#pragma unroll
        for (int s = 0; s < 2; ++s) {
            const int kb = s * 32;
            v16h afr[2], bfr[2];
#pragma unroll
            for (int t = 0; t < 2; ++t) {
                const int mrow = wm * 32 + t * 16 + l15;
                const v8h lo = *(const v8h*)&Ab[mrow * 64 + kb + kbA];
                const v8h hi = *(const v8h*)&Ab[mrow * 64 + kb + 16 + kbA];
#pragma unroll
                for (int e = 0; e < 8; ++e) { afr[t][e] = lo[e]; afr[t][e + 8] = hi[e]; }
            }
#pragma unroll
            for (int t = 0; t < 2; ++t) {
                const int nloc = wn * 32 + t * 16 + l15;
                const v8h lo = *(const v8h*)&Bb[nloc * 64 + kb + kbB];
                const v8h hi = *(const v8h*)&Bb[nloc * 64 + kb + kbB + 16];
#pragma unroll
                for (int e = 0; e < 8; ++e) { bfr[t][e] = lo[e]; bfr[t][e + 8] = hi[e]; }
            }
#pragma unroll
            for (int ti = 0; ti < 2; ++ti)
#pragma unroll
                for (int tj = 0; tj < 2; ++tj)
                    acc[ti * 2 + tj] = __builtin_amdgcn_wmma_f32_16x16x32_f16(
                        false, afr[ti], false, bfr[tj], (short)0, acc[ti * 2 + tj],
                        false, false);
        }
        __syncthreads();
        if (i + 2 < nch) issue((i + 2) << 6, i & 1);
    }

    // --- epilogue: bias (+softplus), f32 store, optional f16 copy ---
#pragma unroll
    for (int ti = 0; ti < 2; ++ti) {
#pragma unroll
        for (int tj = 0; tj < 2; ++tj) {
            const int n_g = n0 + wn * 32 + tj * 16 + l15;
            if (n_g >= N) continue;
            const float bv = bias ? bias[n_g] : 0.0f;
            const int mbase = gm0 + wm * 32 + ti * 16 + ((lane < 16) ? 0 : 8);
#pragma unroll
            for (int r = 0; r < 8; ++r) {
                float v = acc[ti * 2 + tj][r] + bv;
                if (EPI == 1) v = (v > 20.0f) ? v : log1pf(__expf(v));
                C[(size_t)(mbase + r) * ldc + n_g] = v;
                if (Ch) Ch[(size_t)(mbase + r) * ldch + n_g] = (_Float16)v;
            }
        }
    }
}

// -------------------------------------- causal depthwise conv (K=4) + SiLU
__global__ void conv_silu_kernel(const float* __restrict__ xz,
                                 const float* __restrict__ cw,
                                 const float* __restrict__ cb,
                                 float* __restrict__ xi,
                                 _Float16* __restrict__ xi_h) {
    const int idx = blockIdx.x * 256 + threadIdx.x;   // b*L*D + t*D + d
    const int d = idx & (D_INNER - 1);
    const int t = (idx >> 11) & (LL - 1);
    const int b = idx >> 21;
    float acc = cb[d];
#pragma unroll
    for (int j = 0; j < D_CONV; ++j) {
        const int tt = t - (D_CONV - 1) + j;
        if (tt >= 0)
            acc += xz[(size_t)(b * LL + tt) * (2 * D_INNER) + d] * cw[d * D_CONV + j];
    }
    acc = acc / (1.0f + __expf(-acc));                // SiLU
    xi[(size_t)idx]   = acc;
    xi_h[(size_t)idx] = (_Float16)acc;
}

// ------------------------------------------------------- selective scan
__global__ __launch_bounds__(256)
void scan_kernel(const float* __restrict__ xz,       // delta cols [0,2048), z cols [2048,4096)
                 const float* __restrict__ xi_in,
                 const float* __restrict__ dbc,      // [row][96]: 64 dt | 16 B | 16 C
                 const float* __restrict__ A_log,
                 const float* __restrict__ Dp,
                 _Float16* __restrict__ y_h) {       // gated y (f16, feeds out_proj)
    __shared__ float Bc[64 * 16];
    __shared__ float Cc[64 * 16];
    const int tid = threadIdx.x;
    const int b   = blockIdx.x >> 3;
    const int d   = ((blockIdx.x & 7) << 8) + tid;

    float acoef[D_STATE], h[D_STATE];
#pragma unroll
    for (int n = 0; n < D_STATE; ++n) {
        acoef[n] = -__expf(A_log[d * D_STATE + n]);
        h[n] = 0.0f;
    }
    const float Dd = Dp[d];

    for (int t0 = 0; t0 < LL; t0 += 64) {
        {
            const int r = tid >> 2, q = tid & 3;
            const size_t rb = (size_t)(b * LL + t0 + r) * 96;
            const float4 vb = *(const float4*)(dbc + rb + DT_RANK + q * 4);
            const float4 vc = *(const float4*)(dbc + rb + DT_RANK + D_STATE + q * 4);
            ((float4*)&Bc[r * 16])[q] = vb;
            ((float4*)&Cc[r * 16])[q] = vc;
        }
        __syncthreads();
        for (int tt = 0; tt < 64; ++tt) {
            const size_t row = (size_t)(b * LL + t0 + tt);
            const float dv = xz[row * (2 * D_INNER) + d];
            const float zv = xz[row * (2 * D_INNER) + D_INNER + d];
            const float xv = xi_in[row * D_INNER + d];
            float y = 0.0f;
#pragma unroll
            for (int n = 0; n < D_STATE; ++n) {
                const float a  = __expf(dv * acoef[n]);
                const float hn = a * h[n] + dv * Bc[tt * 16 + n] * xv;
                h[n] = hn;
                y += hn * Cc[tt * 16 + n];
            }
            y += Dd * xv;
            const float g = zv / (1.0f + __expf(-zv));
            y_h[row * D_INNER + d] = (_Float16)(y * g);
        }
        __syncthreads();
    }
}

// ---------------------------------------------------------------- launch
extern "C" void kernel_launch(void* const* d_in, const int* in_sizes, int n_in,
                              void* d_out, int out_size, void* d_ws, size_t ws_size,
                              hipStream_t stream) {
    const float* x       = (const float*)d_in[0];
    const float* norm_w  = (const float*)d_in[1];
    const float* norm_b  = (const float*)d_in[2];
    const float* in_w    = (const float*)d_in[3];
    const float* in_b    = (const float*)d_in[4];
    const float* conv_w  = (const float*)d_in[5];
    const float* conv_b  = (const float*)d_in[6];
    const float* xproj_w = (const float*)d_in[7];
    const float* dt_w    = (const float*)d_in[8];
    const float* dt_b    = (const float*)d_in[9];
    const float* A_log   = (const float*)d_in[10];
    const float* Dp      = (const float*)d_in[11];
    const float* out_w   = (const float*)d_in[12];
    const float* out_b   = (const float*)d_in[13];
    float* out = (float*)d_out;

    // -------- workspace layout --------
    float* wsf = (float*)d_ws;
    float* xz  = wsf;                                  // 2048*4096 f32
    float* xi  = xz + (size_t)NROW * 2 * D_INNER;      // 2048*2048 f32
    float* dbc = xi + (size_t)NROW * D_INNER;          // 2048*96   f32
    _Float16* h16     = (_Float16*)(dbc + (size_t)NROW * 96);
    _Float16* xn_h    = h16;                           // 2048*1024
    _Float16* xi_h    = xn_h  + (size_t)NROW * DIM;    // 2048*2048
    _Float16* dbc_h   = xi_h  + (size_t)NROW * D_INNER;// 2048*96
    _Float16* y_h     = dbc_h + (size_t)NROW * 96;     // 2048*2048
    _Float16* in_wT   = y_h   + (size_t)NROW * D_INNER;// [4096][1024]
    _Float16* xp_wT   = in_wT + (size_t)2 * D_INNER * DIM;   // [128][2048]
    _Float16* dt_wT   = xp_wT + (size_t)128 * D_INNER;       // [2048][64]
    _Float16* out_wT  = dt_wT + (size_t)D_INNER * DT_RANK;   // [1024][2048]

    const dim3 tb(32, 8);
    // weights: transpose + convert to f16 [Npad][K]
    transp_f32_to_f16<<<dim3(4096/32, 1024/32), tb, 0, stream>>>(in_w,   DIM,     2*D_INNER, in_wT);
    transp_f32_to_f16<<<dim3(128/32,  2048/32), tb, 0, stream>>>(xproj_w,D_INNER, 96,        xp_wT);
    transp_f32_to_f16<<<dim3(2048/32, 64/32),   tb, 0, stream>>>(dt_w,   DT_RANK, D_INNER,   dt_wT);
    transp_f32_to_f16<<<dim3(1024/32, 2048/32), tb, 0, stream>>>(out_w,  D_INNER, DIM,       out_wT);

    // 1) layernorm -> f16
    ln_kernel<<<NROW, 256, 0, stream>>>(x, norm_w, norm_b, xn_h);
    // 2) in_proj: xz = xn @ in_w + in_b          (2048x1024x4096)
    gemm_wmma<0><<<dim3(4096/64, NROW/128), 256, 0, stream>>>(
        xn_h, DIM, in_wT, DIM, in_b, xz, 2*D_INNER, nullptr, 0,
        NROW, 2*D_INNER, DIM);
    // 3) causal depthwise conv + SiLU -> xi (f32 + f16)
    conv_silu_kernel<<<(NROW * D_INNER)/256, 256, 0, stream>>>(xz, conv_w, conv_b, xi, xi_h);
    // 4) x_proj: dbc = xi @ xproj_w              (2048x2048x96), f16 copy for dt GEMM
    gemm_wmma<0><<<dim3(2, NROW/128), 256, 0, stream>>>(
        xi_h, D_INNER, xp_wT, D_INNER, nullptr, dbc, 96, dbc_h, 96,
        NROW, 96, D_INNER);
    // 5) dt_proj + softplus -> delta over dead xi_raw half of xz (ldc=4096)
    gemm_wmma<1><<<dim3(D_INNER/64, NROW/128), 256, 0, stream>>>(
        dbc_h, 96, dt_wT, DT_RANK, dt_b, xz, 2*D_INNER, nullptr, 0,
        NROW, D_INNER, DT_RANK);
    // 6) selective scan; gated y -> f16
    scan_kernel<<<16, 256, 0, stream>>>(xz, xi, dbc, A_log, Dp, y_h);
    // 7) out_proj: out = ygated @ out_w + out_b  (2048x2048x1024)
    gemm_wmma<0><<<dim3(DIM/64, NROW/128), 256, 0, stream>>>(
        y_h, D_INNER, out_wT, D_INNER, out_b, out, DIM, nullptr, 0,
        NROW, DIM, D_INNER);
}
